// ThinkingLSTM_21440476741883
// MI455X (gfx1250) — compile-verified, matrix operand order checked by reading, not run on hardware
//
#include <hip/hip_runtime.h>
#include <hip/hip_bf16.h>

// ---------------- problem constants ----------------
#define BATCH     128
#define HID       512
#define VOC       512
#define SEQ       256
#define NDEC      253             // SEQ-3
#define STEP_THINK0 SEQ           // 256
#define STEP_DEC0   (SEQ + 9)     // 265
#define NSTEPS      (SEQ + 9 + NDEC) // 518
#define ENDTOK    511
#define MAXTHINK  8

// ---------------- launch shape ----------------
#define NWG  32                   // persistent workgroups (co-resident)
#define TPB  256                  // 8 wave32 per WG

typedef __bf16 v16bf __attribute__((ext_vector_type(16)));
typedef float  v8f   __attribute__((ext_vector_type(8)));
typedef unsigned int u32x4 __attribute__((ext_vector_type(4)));

union BFTile { u32x4 q[2]; v16bf v; };

__device__ __forceinline__ unsigned short f2bf(float f) {
  unsigned int u = __float_as_uint(f);
  u += 0x7fffu + ((u >> 16) & 1u);          // round-to-nearest-even
  return (unsigned short)(u >> 16);
}
__device__ __forceinline__ float sigm(float x) { return 1.0f / (1.0f + __expf(-x)); }

__device__ __forceinline__ float gumbel_hash(unsigned int s) {
  s = (s ^ 61u) ^ (s >> 16); s *= 9u; s ^= s >> 4; s *= 0x27d4eb2du; s ^= s >> 15;
  float u = ((float)(s >> 8) + 0.5f) * (1.0f / 16777216.0f);
  return -__logf(-__logf(u));
}

// A-tile (16x32 bf16) from row-major [*,512] bf16. (ISA 7.12.2 lane layout)
__device__ __forceinline__ v16bf load_a(const unsigned short* __restrict__ base,
                                        int row, int k) {
  const unsigned short* p = base + row * 512 + k;
  BFTile t;
  t.q[0] = *(const u32x4*)(p);
  t.q[1] = *(const u32x4*)(p + 16);
  return t.v;
}
// B-tile (32x16 bf16) from packed layout: tile*512 + lane*16, two b128 loads.
__device__ __forceinline__ v16bf load_b(const unsigned short* __restrict__ packed,
                                        int tile, int lane) {
  const unsigned short* p = packed + tile * 512 + lane * 16;
  BFTile t;
  t.q[0] = *(const u32x4*)(p);
  t.q[1] = *(const u32x4*)(p + 8);
  return t.v;
}

// ---------------- software grid barrier ----------------
__device__ __forceinline__ void gsync(int* bar) {
  __syncthreads();
  if (threadIdx.x == 0) {
    __threadfence();
    int gen = __hip_atomic_load(&bar[1], __ATOMIC_ACQUIRE, __HIP_MEMORY_SCOPE_AGENT);
    int prev = __hip_atomic_fetch_add(&bar[0], 1, __ATOMIC_ACQ_REL, __HIP_MEMORY_SCOPE_AGENT);
    if (prev == NWG - 1) {
      __hip_atomic_store(&bar[0], 0, __ATOMIC_RELAXED, __HIP_MEMORY_SCOPE_AGENT);
      __hip_atomic_fetch_add(&bar[1], 1, __ATOMIC_RELEASE, __HIP_MEMORY_SCOPE_AGENT);
    } else {
      while (__hip_atomic_load(&bar[1], __ATOMIC_ACQUIRE, __HIP_MEMORY_SCOPE_AGENT) == gen)
        __builtin_amdgcn_s_sleep(1);
    }
    __threadfence();
  }
  __syncthreads();
}

// ---------------- init kernel ----------------
__global__ void tl_init(unsigned short* hb, int* active, int* steps, int* tok, int* bar) {
  int i = blockIdx.x * blockDim.x + threadIdx.x;
  if (i < 2 * BATCH * HID) hb[i] = 0;
  if (i < BATCH) { active[i] = 1; steps[i] = 0; tok[i] = 0; }
  if (i < 2) bar[i] = 0;
}

// ---------------- weight packer: f32 [N,512] -> bf16 WMMA B-tile layout ----------
__global__ void tl_pack(const float* __restrict__ Wih, const float* __restrict__ Whh,
                        const float* __restrict__ Wfc,
                        unsigned short* __restrict__ pWih, unsigned short* __restrict__ pWhh,
                        unsigned short* __restrict__ pWfc) {
  const int NIH = 4 * HID * 512;
  const int NFC = VOC * 512;
  int i = blockIdx.x * blockDim.x + threadIdx.x;
  const float* src; unsigned short* dst; int idx;
  if (i < NIH)            { src = Wih; dst = pWih; idx = i; }
  else if (i < 2 * NIH)   { src = Whh; dst = pWhh; idx = i - NIH; }
  else if (i < 2*NIH+NFC) { src = Wfc; dst = pWfc; idx = i - 2 * NIH; }
  else return;
  int e       = idx & 511;
  int tileIdx = idx >> 9;
  int kt = tileIdx & 15;         // 512/32 = 16 K-tiles
  int nt = tileIdx >> 4;
  int lane = e >> 4;
  int rem  = e & 15;
  int v = rem >> 1, p = rem & 1;
  int n  = nt * 16 + (lane & 15);
  int kb = (lane >> 4) * 8;
  int k  = kt * 32 + kb + ((v < 4) ? (2 * v + p) : (16 + 2 * (v - 4) + p));
  dst[idx] = f2bf(src[n * 512 + k]);   // B[k][n] = W[n][k]
}

// ---------------- persistent LSTM kernel ----------------
extern "C" __global__ void __launch_bounds__(TPB, 1)
tl_main(const float* __restrict__ x, const int* __restrict__ lengths,
        const float* __restrict__ b_ih, const float* __restrict__ b_hh,
        const float* __restrict__ b_fc,
        const float* __restrict__ Wih,              // original f32 for one-hot gather
        const unsigned short* __restrict__ pWih, const unsigned short* __restrict__ pWhh,
        const unsigned short* __restrict__ pWfc,
        unsigned short* __restrict__ hb,    // DOUBLE buffered bf16 h mirror [2][B,H]
        unsigned short* __restrict__ h2b,   // unmasked new h, bf16 [B,H]
        unsigned short* __restrict__ ibuf,  // DOUBLE buffered dense input [2][B,V]
        int* tok, int* active, int* steps, int* bar,
        float* __restrict__ out) {
  const int g    = blockIdx.x;       // WG owns hidden cols [16g, 16g+16)
  const int tid  = threadIdx.x;
  const int wave = tid >> 5;         // m-tile for gate GEMM
  const int lane = tid & 31;
  const int l15  = lane & 15;
  const int kb   = (lane >> 4) * 8;
  const int gt   = g * TPB + tid;

  const int j    = g * 16 + l15;                 // owned hidden column
  const int mrb  = wave * 16 + (lane >> 4) * 8;  // first of 8 owned rows

  __shared__ float slog[16 * 512];   // logits tile (32 KB)
  __shared__ float sval[TPB];
  __shared__ int   sidx[TPB];

  // persistent per-thread LSTM state: c and h for the 8 owned (m, j) cells
  float creg[8], hreg[8];
#pragma unroll
  for (int r = 0; r < 8; ++r) { creg[r] = 0.0f; hreg[r] = 0.0f; }

  // -------- prologue: stage x[:,0] into ibuf[0] --------
  for (int e = gt; e < BATCH * VOC; e += NWG * TPB) {
    int b = e >> 9, v = e & 511;
    ibuf[e] = f2bf(x[(size_t)b * SEQ * VOC + v]);
  }
  gsync(bar);

  for (int step = 0; step < NSTEPS; ++step) {
    const bool isEnc   = step < STEP_THINK0;
    const bool isThink = (step >= STEP_THINK0) && (step < STEP_DEC0);
    const bool denseIH = isEnc || (step == STEP_THINK0);
    const int cur = step & 1, nxt = (step + 1) & 1;

    // ================= phase G+U: gates GEMM + register-local cell update ======
    v8f acc[4];                      // jj: 0=i 1=f 2=g 3=o, all for column j
    if (denseIH) {
#pragma unroll
      for (int jj = 0; jj < 4; ++jj) {
        int n = jj * 512 + j;
        float bias = b_ih[n] + b_hh[n];
#pragma unroll
        for (int r = 0; r < 8; ++r) acc[jj][r] = bias;
      }
      const unsigned short* ib = ibuf + (size_t)cur * BATCH * VOC;
      const int arow = wave * 16 + l15;
#pragma unroll 2
      for (int kt = 0; kt < 16; ++kt) {
        v16bf a = load_a(ib, arow, kt * 32 + kb);
        __builtin_prefetch(pWih + ((size_t)g * 16 + kt + 1) * 512 + lane * 16, 0, 1);
#pragma unroll
        for (int jj = 0; jj < 4; ++jj) {
          v16bf bm = load_b(pWih, (g + jj * 32) * 16 + kt, lane);
          acc[jj] = __builtin_amdgcn_wmma_f32_16x16x32_bf16(
                        false, a, false, bm, (short)0, acc[jj], false, false);
        }
      }
    } else {
      // one-hot input: inp @ W_ih^T == W_ih[n, tok[m]]  (exact f32 gather)
      int tokm[8];
#pragma unroll
      for (int r = 0; r < 8; ++r) tokm[r] = tok[mrb + r];
#pragma unroll
      for (int jj = 0; jj < 4; ++jj) {
        int n = jj * 512 + j;
        float bias = b_ih[n] + b_hh[n];
#pragma unroll
        for (int r = 0; r < 8; ++r)
          acc[jj][r] = bias + Wih[(size_t)n * 512 + tokm[r]];
      }
    }
    {
      const unsigned short* hcur = hb + (size_t)cur * BATCH * HID;
      const int arow = wave * 16 + l15;
#pragma unroll 2
      for (int kt = 0; kt < 16; ++kt) {
        v16bf a = load_a(hcur, arow, kt * 32 + kb);
        __builtin_prefetch(pWhh + ((size_t)g * 16 + kt + 1) * 512 + lane * 16, 0, 1);
#pragma unroll
        for (int jj = 0; jj < 4; ++jj) {
          v16bf bm = load_b(pWhh, (g + jj * 32) * 16 + kt, lane);
          acc[jj] = __builtin_amdgcn_wmma_f32_16x16x32_bf16(
                        false, a, false, bm, (short)0, acc[jj], false, false);
        }
      }
    }
    // cell update entirely in registers (acc slot r <-> row m = mrb + r)
    {
      unsigned short* hnxt = hb + (size_t)nxt * BATCH * HID;
#pragma unroll
      for (int r = 0; r < 8; ++r) {
        const int m = mrb + r;
        float cn = sigm(acc[1][r]) * creg[r] + sigm(acc[0][r]) * tanhf(acc[2][r]);
        float hn = sigm(acc[3][r]) * tanhf(cn);
        bool upd = isThink ? (active[m] != 0) : true;
        if (upd) { creg[r] = cn; hreg[r] = hn; }
        if (!isEnc) h2b[m * 512 + j] = f2bf(hn);   // logits use unmasked h2
        hnxt[m * 512 + j] = f2bf(hreg[r]);
      }
    }
    // stage next dense input while GEMM results drain (disjoint buffer)
    if (isEnc) {
      if (step + 1 < SEQ) {
        const size_t xoff = (size_t)(step + 1) * VOC;
        unsigned short* ibn = ibuf + (size_t)nxt * BATCH * VOC;
        for (int e = gt; e < BATCH * VOC; e += NWG * TPB) {
          int b = e >> 9, v = e & 511;
          ibn[e] = f2bf(x[(size_t)b * SEQ * VOC + xoff + v]);
        }
      } else {  // step==255: stage dec_in = x[b, lengths[b]-1] into ibuf[0]
        unsigned short* ibn = ibuf + (size_t)nxt * BATCH * VOC;   // nxt==0
        for (int e = gt; e < BATCH * VOC; e += NWG * TPB) {
          int b = e >> 9, v = e & 511;
          int t = lengths[b] - 1;
          ibn[e] = f2bf(x[(size_t)b * SEQ * VOC + (size_t)t * VOC + v]);
        }
      }
    }
    gsync(bar);

    if (!isEnc) {
      // ========== phase L+S: logits (4x-redundant per m-tile) + local argmax ====
      {
        const int m0L  = (g >> 2) * 16;     // 4 WGs share one 16-row m-tile
        const int arow = m0L + l15;
        v8f lacc[4];
#pragma unroll
        for (int jj = 0; jj < 4; ++jj) {
          int n = (4 * wave + jj) * 16 + l15;
          float bias = b_fc[n];
#pragma unroll
          for (int r = 0; r < 8; ++r) lacc[jj][r] = bias;
        }
#pragma unroll 2
        for (int kt = 0; kt < 16; ++kt) {
          v16bf a = load_a(h2b, arow, kt * 32 + kb);
#pragma unroll
          for (int jj = 0; jj < 4; ++jj) {
            v16bf bm = load_b(pWfc, (4 * wave + jj) * 16 + kt, lane);
            lacc[jj] = __builtin_amdgcn_wmma_f32_16x16x32_bf16(
                           false, a, false, bm, (short)0, lacc[jj], false, false);
          }
        }
        const int lrow0 = (lane >> 4) * 8;
#pragma unroll
        for (int jj = 0; jj < 4; ++jj) {
          int n = (4 * wave + jj) * 16 + l15;
#pragma unroll
          for (int r = 0; r < 8; ++r) slog[(lrow0 + r) * 512 + n] = lacc[jj][r];
        }
      }
      __syncthreads();

      // local argmax + state machine for this WG's 4 rows
      for (int rr = 0; rr < BATCH / NWG; ++rr) {
        const int b    = g * (BATCH / NWG) + rr;
        const int lrow = (g & 3) * 4 + rr;
        float best = -3.4e38f; int bi = 0;
        for (int v = tid; v < VOC; v += TPB) {
          unsigned seed = (unsigned)step * 0x01000193u ^ (unsigned)(b * 521 + v) * 0x9e3779b9u;
          float key = slog[lrow * 512 + v] + gumbel_hash(seed);
          if (key > best) { best = key; bi = v; }
        }
        sval[tid] = best; sidx[tid] = bi;
        __syncthreads();
        for (int off = TPB / 2; off > 0; off >>= 1) {
          if (tid < off) {
            bool take = (sval[tid + off] > sval[tid]) ||
                        (sval[tid + off] == sval[tid] && sidx[tid + off] < sidx[tid]);
            if (take) { sval[tid] = sval[tid + off]; sidx[tid] = sidx[tid + off]; }
          }
          __syncthreads();
        }
        const int sel = sidx[0];
        if (isThink) {
          if (tid == 0) {
            int act = active[b];
            if (act) tok[b] = sel;
            int sn = steps[b] + (act ? 1 : 0);
            steps[b]  = sn;
            active[b] = (act && sel != ENDTOK && sn <= MAXTHINK) ? 1 : 0;
          }
        } else {
          const int tdec = step - STEP_DEC0;
          for (int v = tid; v < VOC - 1; v += TPB)
            out[(size_t)b * NDEC * (VOC - 1) + (size_t)tdec * (VOC - 1) + v] =
                slog[lrow * 512 + v];
          if (tid == 0) tok[b] = sel;
        }
        __syncthreads();
      }
      gsync(bar);
    }
  }

  gsync(bar);
  if (g == 0 && tid < BATCH)
    out[(size_t)BATCH * NDEC * (VOC - 1) + tid] = (float)steps[tid];
}

// ---------------- launcher ----------------
extern "C" void kernel_launch(void* const* d_in, const int* in_sizes, int n_in,
                              void* d_out, int out_size, void* d_ws, size_t ws_size,
                              hipStream_t stream) {
  (void)in_sizes; (void)n_in; (void)out_size; (void)ws_size;
  const float* x    = (const float*)d_in[0];
  const int*   len  = (const int*)  d_in[1];
  const float* Wih  = (const float*)d_in[2];
  const float* Whh  = (const float*)d_in[3];
  const float* bih  = (const float*)d_in[4];
  const float* bhh  = (const float*)d_in[5];
  const float* Wfc  = (const float*)d_in[6];
  const float* bfc  = (const float*)d_in[7];
  float* out = (float*)d_out;

  char* ws = (char*)d_ws;
  size_t off = 0;
  auto alloc = [&](size_t bytes) -> void* {
    void* p = ws + off;
    off = (off + bytes + 255) & ~(size_t)255;
    return p;
  };
  unsigned short* pWih = (unsigned short*)alloc((size_t)4 * HID * 512 * 2);
  unsigned short* pWhh = (unsigned short*)alloc((size_t)4 * HID * 512 * 2);
  unsigned short* pWfc = (unsigned short*)alloc((size_t)VOC * 512 * 2);
  unsigned short* hb   = (unsigned short*)alloc((size_t)2 * BATCH * HID * 2);
  unsigned short* h2b  = (unsigned short*)alloc((size_t)BATCH * HID * 2);
  unsigned short* ibuf = (unsigned short*)alloc((size_t)2 * BATCH * VOC * 2);
  int* tokb   = (int*)alloc(BATCH * 4);
  int* active = (int*)alloc(BATCH * 4);
  int* steps  = (int*)alloc(BATCH * 4);
  int* bar    = (int*)alloc(64);

  tl_init<<<(2 * BATCH * HID + TPB - 1) / TPB, TPB, 0, stream>>>(hb, active, steps, tokb, bar);
  const int packN = 2 * (4 * HID * 512) + VOC * 512;
  tl_pack<<<(packN + TPB - 1) / TPB, TPB, 0, stream>>>(Wih, Whh, Wfc, pWih, pWhh, pWfc);
  tl_main<<<NWG, TPB, 0, stream>>>(x, len, bih, bhh, bfc, Wih, pWih, pWhh, pWfc,
                                   hb, h2b, ibuf, tokb, active, steps, bar, out);
}
